// RealSpikingGNN_16544214024860
// MI455X (gfx1250) — compile-verified, matrix operand order checked by reading, not run on hardware
//
#include <hip/hip_runtime.h>
#include <math.h>

typedef __attribute__((ext_vector_type(16))) __bf16 v16bf;
typedef __attribute__((ext_vector_type(8)))  float  v8f;
typedef __attribute__((ext_vector_type(4)))  float  v4f;

#define N_NODES 16384
#define D_IN    64
#define D_H     128
#define D_H2    32
#define D_OUT   4
#define KCHUNK  32
#define BLOCK_M 64      // rows per workgroup in agg kernel (4 waves * 16 rows)

// ---------------------------------------------------------------------------
// Kernel 1: h = tanh(x @ w_in^T + b_in), stored TRANSPOSED as bf16: hT[d][n]
// One block per node, 128 threads (one per output feature).
// ---------------------------------------------------------------------------
__global__ __launch_bounds__(D_H)
void input_layer_kernel(const float* __restrict__ x,
                        const float* __restrict__ w_in,
                        const float* __restrict__ b_in,
                        __bf16* __restrict__ hT) {
    __shared__ float xs[D_IN];
    const int n = blockIdx.x;
    const int d = threadIdx.x;
    if (d < D_IN) xs[d] = x[(size_t)n * D_IN + d];
    __syncthreads();
    float acc = b_in[d];
    const float* w = w_in + (size_t)d * D_IN;
    #pragma unroll
    for (int k = 0; k < D_IN; ++k) acc = fmaf(xs[k], w[k], acc);
    hT[(size_t)d * N_NODES + n] = (__bf16)tanhf(acc);
}

// ---------------------------------------------------------------------------
// Aggregation kernel: agg[n][d] = sum_k adj[n][k] * f[k][d]   (f given as fT)
// BF16 WMMA 16x16x32, fp32 accumulate. Adjacency is streamed (2 x 1 GiB >>
// 192 MB L2) so its loads are NON-TEMPORAL; fT (4 MB) stays L2-resident and
// is staged through LDS once per workgroup per K-chunk.
//
// A fragment (16x32 bf16): lanes 0-15 hold row M=lane, K=0..15 packed in
// VGPR0..7; lanes 16-31 hold row M=lane-16, K=16..31.  Each lane loads 16
// contiguous fp32 from its adjacency row and converts to bf16 in registers
// (0/1 exact in bf16).
// B fragment (32x16 bf16): lanes 0-15 hold col N=lane, K=0..15; lanes 16-31
// hold col N=lane-16, K=16..31 -> each lane reads 16 contiguous bf16 from the
// LDS-staged fT slab.
// ---------------------------------------------------------------------------
__global__ __launch_bounds__(128)
void agg_kernel(const float*  __restrict__ adj,
                const __bf16* __restrict__ fT,
                float* __restrict__ agg,
                float* __restrict__ deg,
                int compute_deg) {
    __shared__ __align__(32) __bf16 bsh[D_H * KCHUNK];   // [col][k], 8 KB
    __shared__ float degsh[128];

    const int tid  = threadIdx.x;
    const int wave = tid >> 5;          // 0..3
    const int lane = tid & 31;
    const int lane15 = lane & 15;
    const int koff   = (lane >> 4) * 16;   // 0 or 16

    const int row_base = blockIdx.x * BLOCK_M + wave * 16;
    const int arow     = row_base + lane15;
    const float* aptr  = adj + (size_t)arow * N_NODES + koff;

    v8f acc[8] = {};
    float degsum = 0.0f;

    for (int k0 = 0; k0 < N_NODES; k0 += KCHUNK) {
        // ---- stage fT[c][k0 .. k0+31] into LDS (each thread: one column) ----
        {
            const int c = tid;  // 0..127
            const __bf16* src = fT + (size_t)c * N_NODES + k0;
            uint4* dst = (uint4*)&bsh[c * KCHUNK];
            dst[0] = ((const uint4*)src)[0];
            dst[1] = ((const uint4*)src)[1];
            dst[2] = ((const uint4*)src)[2];
            dst[3] = ((const uint4*)src)[3];
        }
        __syncthreads();

        // ---- A fragment: 16 contiguous fp32, streamed non-temporally ----
        const v4f* ap = (const v4f*)(aptr + k0);
        const v4f a0 = __builtin_nontemporal_load(ap + 0);
        const v4f a1 = __builtin_nontemporal_load(ap + 1);
        const v4f a2 = __builtin_nontemporal_load(ap + 2);
        const v4f a3 = __builtin_nontemporal_load(ap + 3);
        __builtin_prefetch(aptr + k0 + KCHUNK, 0, 0);

        if (compute_deg) {
            degsum += a0[0] + a0[1] + a0[2] + a0[3];
            degsum += a1[0] + a1[1] + a1[2] + a1[3];
            degsum += a2[0] + a2[1] + a2[2] + a2[3];
            degsum += a3[0] + a3[1] + a3[2] + a3[3];
        }

        v16bf afrag;
        afrag[0]  = (__bf16)a0[0];  afrag[1]  = (__bf16)a0[1];
        afrag[2]  = (__bf16)a0[2];  afrag[3]  = (__bf16)a0[3];
        afrag[4]  = (__bf16)a1[0];  afrag[5]  = (__bf16)a1[1];
        afrag[6]  = (__bf16)a1[2];  afrag[7]  = (__bf16)a1[3];
        afrag[8]  = (__bf16)a2[0];  afrag[9]  = (__bf16)a2[1];
        afrag[10] = (__bf16)a2[2];  afrag[11] = (__bf16)a2[3];
        afrag[12] = (__bf16)a3[0];  afrag[13] = (__bf16)a3[1];
        afrag[14] = (__bf16)a3[2];  afrag[15] = (__bf16)a3[3];

        // ---- 8 column tiles of 16: WMMA ----
        #pragma unroll
        for (int j = 0; j < 8; ++j) {
            const v16bf bfrag =
                *(const v16bf*)&bsh[(size_t)(j * 16 + lane15) * KCHUNK + koff];
            acc[j] = __builtin_amdgcn_wmma_f32_16x16x32_bf16(
                false, afrag, false, bfrag, (short)0, acc[j], false, false);
        }
        __syncthreads();
    }

    // ---- store accumulators: C layout: VGPR r -> M = r + (lane>=16 ? 8:0) ----
    const int mhi = (lane >> 4) * 8;
    #pragma unroll
    for (int j = 0; j < 8; ++j) {
        const int col = j * 16 + lane15;
        #pragma unroll
        for (int r = 0; r < 8; ++r) {
            const int row = row_base + mhi + r;
            agg[(size_t)row * D_H + col] = acc[j][r];
        }
    }

    if (compute_deg) {
        degsh[tid] = degsum;
        __syncthreads();
        if (lane < 16) {
            deg[row_base + lane] = degsh[wave * 32 + lane] + degsh[wave * 32 + lane + 16];
        }
    }
}

// ---------------------------------------------------------------------------
// Spike layer 1: m = deg>0 ? agg/deg : h ;  s1 = ((m @ w1^T + b1)/tau >= 1)
// One block per node, 128 threads. Output transposed bf16 for next agg pass.
// ---------------------------------------------------------------------------
__global__ __launch_bounds__(D_H)
void spike1_kernel(const float* __restrict__ agg,
                   const float* __restrict__ deg,
                   const __bf16* __restrict__ hT,
                   const float* __restrict__ w1,
                   const float* __restrict__ b1,
                   __bf16* __restrict__ s1T) {
    __shared__ float m[D_H];
    const int n = blockIdx.x;
    const int j = threadIdx.x;
    const float dg = deg[n];
    const float v = (dg > 0.0f) ? agg[(size_t)n * D_H + j] / dg
                                : (float)hT[(size_t)j * N_NODES + n];
    m[j] = v;
    __syncthreads();
    float a = b1[j];
    const float* w = w1 + (size_t)j * D_H;
    #pragma unroll 8
    for (int k = 0; k < D_H; ++k) a = fmaf(m[k], w[k], a);
    a *= 0.5f;  // / TAU
    s1T[(size_t)j * N_NODES + n] = (__bf16)((a >= 1.0f) ? 1.0f : 0.0f);
}

// ---------------------------------------------------------------------------
// Spike layer 2 + output head, fused. One block per node, 128 threads.
// ---------------------------------------------------------------------------
__global__ __launch_bounds__(D_H)
void spike2_out_kernel(const float* __restrict__ agg,
                       const float* __restrict__ deg,
                       const __bf16* __restrict__ s1T,
                       const float* __restrict__ w2,
                       const float* __restrict__ b2,
                       const float* __restrict__ w_out,
                       const float* __restrict__ b_out,
                       float* __restrict__ out) {
    __shared__ float m[D_H];
    __shared__ float s2[D_H2];
    const int n = blockIdx.x;
    const int j = threadIdx.x;
    const float dg = deg[n];
    const float v = (dg > 0.0f) ? agg[(size_t)n * D_H + j] / dg
                                : (float)s1T[(size_t)j * N_NODES + n];
    m[j] = v;
    __syncthreads();
    if (j < D_H2) {
        float a = b2[j];
        const float* w = w2 + (size_t)j * D_H;
        #pragma unroll 8
        for (int k = 0; k < D_H; ++k) a = fmaf(m[k], w[k], a);
        s2[j] = ((a * 0.5f) >= 1.0f) ? 1.0f : 0.0f;
    }
    __syncthreads();
    if (j < D_OUT) {
        float a = b_out[j];
        const float* w = w_out + (size_t)j * D_H2;
        #pragma unroll
        for (int k = 0; k < D_H2; ++k) a = fmaf(s2[k], w[k], a);
        out[(size_t)n * D_OUT + j] = a;
    }
}

// ---------------------------------------------------------------------------
extern "C" void kernel_launch(void* const* d_in, const int* in_sizes, int n_in,
                              void* d_out, int out_size, void* d_ws, size_t ws_size,
                              hipStream_t stream) {
    const float* x     = (const float*)d_in[0];
    const float* adj   = (const float*)d_in[1];
    const float* w_in  = (const float*)d_in[2];
    const float* b_in  = (const float*)d_in[3];
    const float* w1    = (const float*)d_in[4];
    const float* b1    = (const float*)d_in[5];
    const float* w2    = (const float*)d_in[6];
    const float* b2    = (const float*)d_in[7];
    const float* w_out = (const float*)d_in[8];
    const float* b_out = (const float*)d_in[9];
    float* out = (float*)d_out;

    char* ws = (char*)d_ws;
    const size_t featT_bytes = (size_t)D_H * N_NODES * sizeof(__bf16);   // 4 MB
    __bf16* hT  = (__bf16*)ws;
    __bf16* s1T = (__bf16*)(ws + featT_bytes);
    float*  agg = (float*)(ws + 2 * featT_bytes);                        // 8 MB
    float*  deg = (float*)(ws + 2 * featT_bytes +
                           (size_t)N_NODES * D_H * sizeof(float));       // 64 KB

    // Layer 0: features (transposed bf16)
    input_layer_kernel<<<N_NODES, D_H, 0, stream>>>(x, w_in, b_in, hT);

    // Pass 1: adj @ h  (also computes degrees), then LIF layer 1
    agg_kernel<<<N_NODES / BLOCK_M, 128, 0, stream>>>(adj, hT, agg, deg, 1);
    spike1_kernel<<<N_NODES, D_H, 0, stream>>>(agg, deg, hT, w1, b1, s1T);

    // Pass 2: adj @ s1, then LIF layer 2 + output head
    agg_kernel<<<N_NODES / BLOCK_M, 128, 0, stream>>>(adj, s1T, agg, deg, 0);
    spike2_out_kernel<<<N_NODES, D_H, 0, stream>>>(agg, deg, s1T, w2, b2,
                                                   w_out, b_out, out);
}
